// TimeAwareCondConv2d_36507222016336
// MI455X (gfx1250) — matrix operand order repaired and checked by reading, not compiled, not run on hardware
//
#include <hip/hip_runtime.h>

typedef __bf16 bf16;
typedef __attribute__((ext_vector_type(16))) __bf16 v16bf;
typedef __attribute__((ext_vector_type(8)))  __bf16 v8bf;
typedef __attribute__((ext_vector_type(8)))  float  v8f;

union FragBF { v16bf v; v8bf h[2]; };

// Problem constants
#define BATCH 16
#define CH    256
#define OCH   256
#define HT    64
#define WD    64
#define HW    4096
#define EXP   8
#define TDIM  512
#define HID   64
#define TAPS  9
#define HP    66          // halo-padded spatial dim
#define HP2   (HP * HP)   // 4356

// ---------------- Kernel 1: global average pool ----------------
__global__ void k_pool(const float* __restrict__ x, float* __restrict__ pooled) {
    const int bc = blockIdx.x;
    const float* p = x + (size_t)bc * HW;
    float s = 0.f;
    for (int i = threadIdx.x; i < HW; i += 256) s += p[i];
    __shared__ float red[256];
    red[threadIdx.x] = s;
    __syncthreads();
    for (int st = 128; st > 0; st >>= 1) {
        if (threadIdx.x < st) red[threadIdx.x] += red[threadIdx.x + st];
        __syncthreads();
    }
    if (threadIdx.x == 0) pooled[bc] = red[0] * (1.0f / HW);
}

// ---------------- Kernel 2a: NCHW f32 -> halo-padded NHWC bf16 (LDS-tiled transpose) ----
// grid = B * H * 4 blocks (one h-row, 64-channel chunk each), 256 threads.
__global__ void k_hwc(const float* __restrict__ x, bf16* __restrict__ xh) {
    __shared__ float tile[64][65];             // padded: conflict-free column reads
    const int blk = blockIdx.x;                // b*256 + h*4 + cc
    const int cc = blk & 3;
    const int h  = (blk >> 2) & 63;
    const int b  = blk >> 8;
    const int c0 = cc * 64;

    // read phase: fully coalesced rows of x
    const int w  = threadIdx.x & 63;
    const int cl = threadIdx.x >> 6;           // 0..3
    const float* src = x + ((size_t)(b * CH + c0) * HW) + h * WD + w;
#pragma unroll
    for (int i = 0; i < 16; ++i)
        tile[cl + 4 * i][w] = src[(size_t)(cl + 4 * i) * HW];
    __syncthreads();

    // write phase: channels contiguous in NHWC output
    const int c2 = threadIdx.x & 63;
    const int w2 = threadIdx.x >> 6;           // 0..3
    bf16* dst = xh + ((size_t)((b * HP + h + 1) * HP) + 1) * CH + c0 + c2;
#pragma unroll
    for (int i = 0; i < 16; ++i)
        dst[(size_t)(w2 + 4 * i) * CH] = (bf16)tile[c2][w2 + 4 * i];
}

// ---------------- Kernel 2b: zero the halo ring ----------------
// grid = B * 260 blocks (border pixels), 256 threads (channels)
__global__ void k_halo(bf16* __restrict__ xh) {
    const int blk = blockIdx.x;
    const int b = blk / 260;
    const int i = blk - b * 260;
    int hp, wp;
    if (i < 66)       { hp = 0;            wp = i; }
    else if (i < 132) { hp = HP - 1;       wp = i - 66; }
    else if (i < 196) { hp = i - 132 + 1;  wp = 0; }
    else              { hp = i - 196 + 1;  wp = HP - 1; }
    xh[((size_t)(b * HP2) + hp * HP + wp) * CH + threadIdx.x] = (bf16)0.f;
}

// ---------------- Kernel 3: router MLP -> rw, b_eff ----------------
__global__ void k_router(const float* __restrict__ te,
                         const float* __restrict__ Wq, const float* __restrict__ bq,
                         const float* __restrict__ Wk, const float* __restrict__ bk,
                         const float* __restrict__ Wv, const float* __restrict__ bv,
                         const float* __restrict__ Wm1, const float* __restrict__ bm1,
                         const float* __restrict__ Wm2, const float* __restrict__ bm2,
                         const float* __restrict__ Wc, const float* __restrict__ bc,
                         const float* __restrict__ eb,
                         const float* __restrict__ pooled,
                         float* __restrict__ rw, float* __restrict__ beff) {
    const int b = blockIdx.x;
    const int h = threadIdx.x;            // 0..63
    __shared__ float xa[HID], m1[HID], xm[HID], logits[EXP], rwL[EXP];

    const float* teb = te + b * TDIM;
    const float* pb  = pooled + b * CH;

    float q = bq[h];
    for (int t = 0; t < TDIM; ++t) q += teb[t] * Wq[t * HID + h];
    float kk = bk[h], vv = bv[h];
    for (int c = 0; c < CH; ++c) { const float p = pb[c]; kk += p * Wk[c * HID + h]; vv += p * Wv[c * HID + h]; }
    const float attn = 1.0f / (1.0f + __expf(-(q * kk)));
    xa[h] = vv * attn;
    __syncthreads();

    float s1 = bm1[h];
    for (int j = 0; j < HID; ++j) s1 += xa[j] * Wm1[j * HID + h];
    m1[h] = s1 / (1.0f + __expf(-s1));    // silu
    __syncthreads();

    float s2 = bm2[h];
    for (int j = 0; j < HID; ++j) s2 += m1[j] * Wm2[j * HID + h];
    xm[h] = xa[h] + s2;
    __syncthreads();

    if (h < EXP) {
        float l = bc[h];
        for (int j = 0; j < HID; ++j) l += xm[j] * Wc[j * EXP + h];
        logits[h] = l;
    }
    __syncthreads();
    if (h == 0) {
        float mx = logits[0];
        for (int e = 1; e < EXP; ++e) mx = fmaxf(mx, logits[e]);
        float sum = 0.f;
        for (int e = 0; e < EXP; ++e) { rwL[e] = __expf(logits[e] - mx); sum += rwL[e]; }
        const float inv = 1.0f / sum;
        for (int e = 0; e < EXP; ++e) { rwL[e] *= inv; rw[b * EXP + e] = rwL[e]; }
    }
    __syncthreads();
    for (int o = h; o < OCH; o += HID) {
        float s3 = 0.f;
        for (int e = 0; e < EXP; ++e) s3 += rwL[e] * eb[e * OCH + o];
        beff[b * OCH + o] = s3;
    }
}

// ---------------- Kernel 4: expert mix + re-layout to [b][tap][o][c] bf16 ----------------
// grid = B * O blocks, 256 threads (one per input channel); contiguous reads and writes.
__global__ void k_mix(const float* __restrict__ ew, const float* __restrict__ rw,
                      bf16* __restrict__ wmix) {
    const int blk = blockIdx.x;               // b*256 + o
    const int o = blk & 255;
    const int b = blk >> 8;
    const int c = threadIdx.x;

    float s[TAPS];
#pragma unroll
    for (int p = 0; p < TAPS; ++p) s[p] = 0.f;

    const float* pw = ew + ((size_t)o * CH + c) * TAPS;   // thread's 9-float span
    const float* r  = rw + b * EXP;
    for (int e = 0; e < EXP; ++e) {
        const float we = r[e];
        const float* q = pw + (size_t)e * ((size_t)OCH * CH * TAPS);
#pragma unroll
        for (int p = 0; p < TAPS; ++p) s[p] += we * q[p];
    }
#pragma unroll
    for (int p = 0; p < TAPS; ++p)
        wmix[((size_t)((b * TAPS + p) * OCH + o)) * CH + c] = (bf16)s[p];
}

// ---------------- Kernel 5: implicit-GEMM conv, WMMA bf16 + async-to-LDS pipeline ------
// grid = (HW/256, O/128, B), 256 threads (8 waves, 2x4).
// Block tile 128(M) x 256(N); wave tile 64x64 -> 4x4 WMMA accumulators.
// Double-buffered LDS staged with global_load_async_to_lds_b128 (ASYNCcnt),
// running global pointers + immediate offsets (INST_OFFSET shifts LDS and global alike).
__global__ __launch_bounds__(256) void k_conv(const bf16* __restrict__ xh,
                                              const bf16* __restrict__ wmix,
                                              const float* __restrict__ beff,
                                              float* __restrict__ out) {
    constexpr int LDA   = 40;                // padded k-stride (elements) to spread LDS banks
    constexpr int ABUFB = 128 * LDA * 2;     // A buffer bytes (10240)
    constexpr int BBUFB = 256 * LDA * 2;     // B buffer bytes (20480)
    __shared__ __align__(16) bf16 Alds[2][128 * LDA];
    __shared__ __align__(16) bf16 Blds[2][256 * LDA];

    const int tid  = threadIdx.x;
    const int b    = blockIdx.z;
    const int m0   = blockIdx.y * 128;
    const int n0   = blockIdx.x * 256;
    const int wave = tid >> 5, lane = tid & 31;
    const int mi   = wave >> 2, ni = wave & 3;    // 2 x 4 wave grid
    const int lm   = lane & 15, kh2 = lane >> 4;  // row-in-tile, K half-group

    v8f acc[4][4];
    const v8f vzero = {};
#pragma unroll
    for (int i = 0; i < 4; ++i)
#pragma unroll
        for (int j = 0; j < 4; ++j) acc[i][j] = vzero;

    // staging: A -> each thread owns (row, 16-wide K chunk) = 32B; B -> full 32-K row = 64B
    const int srowA = tid >> 1;                   // 0..127
    const int skA   = (tid & 1) * 16;             // 0 or 16
    const int gcol  = n0 + tid;                   // B staging pixel (srowB = tid)
    const int hh = gcol >> 6, ww = gcol & 63;

    // Running per-thread global pointers (start at tap 0, c0 = 0)
    const bf16* ga = wmix + ((size_t)(b * TAPS * OCH) + (m0 + srowA)) * CH + skA;
    const bf16* gb = xh + ((size_t)((b * HP + hh) * HP + ww)) * CH;
    const uint32_t lA0 = (uint32_t)(uintptr_t)(&Alds[0][0]) + (uint32_t)(srowA * LDA + skA) * 2u;
    const uint32_t lB0 = (uint32_t)(uintptr_t)(&Blds[0][0]) + (uint32_t)(tid * LDA) * 2u;

    int pm3 = 0;                                  // p % 3 tracker for B tap-cross delta

    auto issue = [&](int bufSel) {
        const uint32_t la = lA0 + (uint32_t)(bufSel * ABUFB);
        const uint32_t lb = lB0 + (uint32_t)(bufSel * BBUFB);
        asm volatile("global_load_async_to_lds_b128 %0, %1, off"           :: "v"(la), "v"(ga) : "memory");
        asm volatile("global_load_async_to_lds_b128 %0, %1, off offset:16" :: "v"(la), "v"(ga) : "memory");
        asm volatile("global_load_async_to_lds_b128 %0, %1, off"           :: "v"(lb), "v"(gb) : "memory");
        asm volatile("global_load_async_to_lds_b128 %0, %1, off offset:16" :: "v"(lb), "v"(gb) : "memory");
        asm volatile("global_load_async_to_lds_b128 %0, %1, off offset:32" :: "v"(lb), "v"(gb) : "memory");
        asm volatile("global_load_async_to_lds_b128 %0, %1, off offset:48" :: "v"(lb), "v"(gb) : "memory");
    };
    auto advance = [&](int kk) {                  // move pointers from step kk to kk+1
        // address VGPRs feed in-flight asm VMEM ops; wait for translations first
        asm volatile("s_wait_xcnt 0x0" ::: "memory");
        if ((kk & 7) == 7) {                      // tap boundary
            ga += OCH * CH - 224;                 // next p, c0=0
            gb += ((pm3 < 2) ? 256 : (HP - 2) * CH) - 224;
            pm3 = (pm3 == 2) ? 0 : pm3 + 1;
        } else {                                  // next 32 channels
            ga += 32;
            gb += 32;
        }
    };

    issue(0);                                     // prime the pipeline
    advance(0);

#pragma unroll 2
    for (int kk = 0; kk < 72; ++kk) {             // K = 9 taps * 256 ch / 32
        const int cur = kk & 1;
        if (kk + 1 < 72) {
            issue(cur ^ 1);
            advance(kk + 1);
            // 12 outstanding, in-order completion: <=6 left ==> current tile landed
            asm volatile("s_wait_asynccnt 0x6" ::: "memory");
        } else {
            asm volatile("s_wait_asynccnt 0x0" ::: "memory");
        }
        __syncthreads();

        // ---- fragment loads (per ISA 16-bit A/B lane layouts) ----
        FragBF a[4], bb[4];
#pragma unroll
        for (int sm = 0; sm < 4; ++sm) {
            const int row = mi * 64 + sm * 16 + lm;
            a[sm].h[0] = *(const v8bf*)&Alds[cur][row * LDA + kh2 * 8];        // K 0..7 / 8..15
            a[sm].h[1] = *(const v8bf*)&Alds[cur][row * LDA + 16 + kh2 * 8];   // K 16..23 / 24..31
        }
#pragma unroll
        for (int sn = 0; sn < 4; ++sn) {
            const int col = ni * 64 + sn * 16 + lm;
            bb[sn].h[0] = *(const v8bf*)&Blds[cur][col * LDA + kh2 * 8];       // K 0..7 / 8..15
            bb[sn].h[1] = *(const v8bf*)&Blds[cur][col * LDA + 16 + kh2 * 8];  // K 16..23 / 24..31
        }
#pragma unroll
        for (int sm = 0; sm < 4; ++sm)
#pragma unroll
            for (int sn = 0; sn < 4; ++sn)
                acc[sm][sn] = __builtin_amdgcn_wmma_f32_16x16x32_bf16(
                    false, a[sm].v, false, bb[sn].v, (short)0, acc[sm][sn], false, false);
        __syncthreads();
    }

    // ---- epilogue: C/D layout: VGPR r -> M = r (+8 for lanes 16..31), N = lane%16 ----
#pragma unroll
    for (int sm = 0; sm < 4; ++sm) {
#pragma unroll
        for (int sn = 0; sn < 4; ++sn) {
            const int obase = m0 + mi * 64 + sm * 16 + kh2 * 8;
            const int pix   = n0 + ni * 64 + sn * 16 + lm;
#pragma unroll
            for (int r = 0; r < 8; ++r) {
                const int o = obase + r;
                out[(((size_t)(b * OCH + o)) << 12) + pix] = acc[sm][sn][r] + beff[b * OCH + o];
            }
        }
    }
}

extern "C" void kernel_launch(void* const* d_in, const int* in_sizes, int n_in,
                              void* d_out, int out_size, void* d_ws, size_t ws_size,
                              hipStream_t stream) {
    (void)in_sizes; (void)n_in; (void)out_size; (void)ws_size;
    const float* x    = (const float*)d_in[0];
    const float* te   = (const float*)d_in[1];
    const float* Wq   = (const float*)d_in[2];
    const float* bq   = (const float*)d_in[3];
    const float* Wk   = (const float*)d_in[4];
    const float* bk   = (const float*)d_in[5];
    const float* Wv   = (const float*)d_in[6];
    const float* bv   = (const float*)d_in[7];
    const float* Wm1  = (const float*)d_in[8];
    const float* bm1  = (const float*)d_in[9];
    const float* Wm2  = (const float*)d_in[10];
    const float* bm2  = (const float*)d_in[11];
    const float* Wc   = (const float*)d_in[12];
    const float* bc   = (const float*)d_in[13];
    const float* ew   = (const float*)d_in[14];
    const float* eb   = (const float*)d_in[15];
    float* out = (float*)d_out;

    // workspace layout
    char* ws = (char*)d_ws;
    float* pooled = (float*)(ws);                         // 16 KB
    float* rw     = (float*)(ws + 16384);                 // 512 B
    float* beff   = (float*)(ws + 16896);                 // 16 KB
    bf16*  xh     = (bf16*)(ws + 65536);                  // 35.68 MB  (B,66,66,C) bf16, halo-padded
    bf16*  wmix   = (bf16*)(ws + 65536 + (size_t)BATCH * HP2 * CH * 2);  // 18.87 MB (B,9,O,C) bf16

    k_pool<<<BATCH * CH, 256, 0, stream>>>(x, pooled);
    k_halo<<<BATCH * 260, 256, 0, stream>>>(xh);
    k_hwc<<<BATCH * HT * 4, 256, 0, stream>>>(x, xh);
    k_router<<<BATCH, HID, 0, stream>>>(te, Wq, bq, Wk, bk, Wv, bv,
                                        Wm1, bm1, Wm2, bm2, Wc, bc, eb,
                                        pooled, rw, beff);
    k_mix<<<BATCH * OCH, 256, 0, stream>>>(ew, rw, wmix);

    dim3 grid(HW / 256, OCH / 128, BATCH);
    k_conv<<<grid, 256, 0, stream>>>(xh, wmix, beff, out);
}
